// EntropySmoothnessLoss_13846974562573
// MI455X (gfx1250) — matrix operand order repaired and checked by reading, not compiled, not run on hardware
//
#include <hip/hip_runtime.h>

// ---------------- problem constants (match reference) ----------------
#define Bn   4
#define H4c  96
#define W4c  312
#define Cc   384
#define MAXD 48
#define DP1  49
#define ENT_Tc   0.1f
#define ENT_THRc 0.6f
#define WINH 12
#define EPSF 1e-8f
// ROI (u0,u1,v0,v1)=(0.2,0.7,2/3,1.0) under python double math:
#define ROI_X0 62
#define ROI_X1 218
#define ROI_Y0 64
#define ROI_Y1 95

#define NEG_INF (-__builtin_inff())

// LDS staging of the full featR row in f16.
// Column stride padded to 400 halves (200 words; 200 % 64 == 8) so the
// per-lane 32B B-fragment reads hit the 2-cycle LDS minimum (no 16-way
// conflicts that a 192-word stride would cause).
#define LDS_STRIDE 400
#define LDS_BYTES  (W4c * LDS_STRIDE * 2)   // 249,600 B  (<= 320KB/WGP)

typedef __attribute__((ext_vector_type(16))) _Float16 v16h;
typedef __attribute__((ext_vector_type(8)))  _Float16 v8h;
typedef __attribute__((ext_vector_type(8)))  float    v8f;

// ============================================================================
// Kernel 1: banded cost volume via v_wmma_f32_16x16x32_f16.
// One workgroup (8 waves) per (b,h) row: featR row staged once into LDS (f16),
// then each wave computes 2-3 x-tiles; 4 accumulators per tile cover the
// 49-wide disparity band x' in [x0-48, x0+15].
// ============================================================================
__device__ inline v16h pack16(float4 f0, float4 f1, float4 f2, float4 f3) {
  v16h r;
  r[0]=(_Float16)f0.x;  r[1]=(_Float16)f0.y;  r[2]=(_Float16)f0.z;  r[3]=(_Float16)f0.w;
  r[4]=(_Float16)f1.x;  r[5]=(_Float16)f1.y;  r[6]=(_Float16)f1.z;  r[7]=(_Float16)f1.w;
  r[8]=(_Float16)f2.x;  r[9]=(_Float16)f2.y;  r[10]=(_Float16)f2.z; r[11]=(_Float16)f2.w;
  r[12]=(_Float16)f3.x; r[13]=(_Float16)f3.y; r[14]=(_Float16)f3.z; r[15]=(_Float16)f3.w;
  return r;
}

// A-matrix (16x32 f16): lanes 0-15 -> M=lane, K halves 0..15 =
// koff..koff+15 (koff = 0 / 16 per half-wave).
__device__ inline v16h load_a(const float* p) {
  const float4* q0 = reinterpret_cast<const float4*>(p);
  const float4* q1 = reinterpret_cast<const float4*>(p + 8);
  return pack16(q0[0], q0[1], q1[0], q1[1]);
}

__device__ inline void store_tile(const v8f& acc, int t, int x0, int N, bool hi,
                                  int b, int h, float* __restrict__ E0) {
  const int xb = x0 - 48 + 16 * t;
  const int xp = xb + N;                    // x' column of this lane
#pragma unroll
  for (int r = 0; r < 8; ++r) {
    const int M = r + (hi ? 8 : 0);
    const int x = x0 + M;
    const int d = x - xp;
    if (x < W4c && d >= 0 && d <= MAXD) {
      const float v = (xp >= 0) ? acc[r] : NEG_INF;   // reference: -inf for x<d
      E0[(((size_t)b * DP1 + d) * H4c + h) * (size_t)W4c + x] = v;
    }
  }
}

__global__ __launch_bounds__(256)
void cv_wmma_row_kernel(const float* __restrict__ fL, const float* __restrict__ fR,
                        float* __restrict__ E0) {
  extern __shared__ _Float16 ldsR[];        // [W4c][LDS_STRIDE] halves

  const int h = blockIdx.x;
  const int b = blockIdx.y;
  const int tid = threadIdx.x;
  const size_t rowbase = (size_t)(b * H4c + h) * W4c;
  const float* fRrow = fR + rowbase * Cc;

  // ---------- Phase 1: stage the whole featR row into LDS as f16 ----------
  // 8-float segments: seg -> (col = seg/48, s = seg%48), K = s*8 .. s*8+7
  const int NSEG = W4c * (Cc / 8);
  for (int seg = tid; seg < NSEG; seg += 256) {
    const int col = seg / (Cc / 8);
    const int s   = seg % (Cc / 8);
    const float4* q = reinterpret_cast<const float4*>(fRrow + (size_t)col * Cc + s * 8);
    float4 f0 = q[0], f1 = q[1];
    v8h hsv;
    hsv[0]=(_Float16)f0.x; hsv[1]=(_Float16)f0.y; hsv[2]=(_Float16)f0.z; hsv[3]=(_Float16)f0.w;
    hsv[4]=(_Float16)f1.x; hsv[5]=(_Float16)f1.y; hsv[6]=(_Float16)f1.z; hsv[7]=(_Float16)f1.w;
    *reinterpret_cast<v8h*>(ldsR + (size_t)col * LDS_STRIDE + s * 8) = hsv;
  }
  __syncthreads();

  // ---------- Phase 2: per-wave x-tiles ----------
  const int l  = tid & 31;
  const int w  = tid >> 5;                  // wave id 0..7
  const int n  = l & 15;
  const bool hi = (l >= 16);

#pragma unroll
  for (int j = 0; j < 3; ++j) {
    const int m = w + 8 * j;                // x-tile index
    if (m >= (W4c + 15) / 16) break;        // 20 tiles
    const int x0 = m * 16;

    // A row pointer (featL), row clamped (out-of-range x masked at store time)
    int arow = x0 + n; if (arow > W4c - 1) arow = W4c - 1;
    const float* pA = fL + (rowbase + (size_t)arow) * Cc + (hi ? 16 : 0);

    // B-fragment LDS bases for the 4 x'-tiles (B layout: lane = N; lanes
    // 16-31 carry K+16..31 -> 16 consecutive halves per lane).
    int lbase[4];
#pragma unroll
    for (int t = 0; t < 4; ++t) {
      int c = x0 - 48 + 16 * t + n;
      int cc = c < 0 ? 0 : (c > W4c - 1 ? W4c - 1 : c);
      lbase[t] = cc * LDS_STRIDE + (hi ? 16 : 0);
    }

    v8f a0 = {0.f,0.f,0.f,0.f,0.f,0.f,0.f,0.f};
    v8f a1 = a0, a2 = a0, a3 = a0;

#pragma unroll 2
    for (int kc = 0; kc < Cc; kc += 32) {
      const v16h A  = load_a(pA + kc);
      const v16h B0 = *reinterpret_cast<const v16h*>(ldsR + lbase[0] + kc);
      const v16h B1 = *reinterpret_cast<const v16h*>(ldsR + lbase[1] + kc);
      const v16h B2 = *reinterpret_cast<const v16h*>(ldsR + lbase[2] + kc);
      const v16h B3 = *reinterpret_cast<const v16h*>(ldsR + lbase[3] + kc);
      a0 = __builtin_amdgcn_wmma_f32_16x16x32_f16(false, A, false, B0, (short)0, a0, false, false);
      a1 = __builtin_amdgcn_wmma_f32_16x16x32_f16(false, A, false, B1, (short)0, a1, false, false);
      a2 = __builtin_amdgcn_wmma_f32_16x16x32_f16(false, A, false, B2, (short)0, a2, false, false);
      a3 = __builtin_amdgcn_wmma_f32_16x16x32_f16(false, A, false, B3, (short)0, a3, false, false);
    }

    store_tile(a0, 0, x0, n, hi, b, h, E0);
    store_tile(a1, 1, x0, n, hi, b, h, E0);
    store_tile(a2, 2, x0, n, hi, b, h, E0);
    store_tile(a3, 3, x0, n, hi, b, h, E0);
  }
}

// ============================================================================
// Entropy helpers (match reference semantics incl. EPS floor on ALL 49 probs)
// ============================================================================
__device__ inline float entropy_finish(float m, int cnt, float Z,
                                       const float* col, size_t stride) {
  float ent = 0.f;
  for (int d = 0; d < DP1; ++d) {
    float v = col[d * stride];
    float p = __expf((v - m) * (1.0f / ENT_Tc)) / Z;   // exp(-inf)=0 for masked
    p = fmaxf(p, EPSF);
    ent -= p * __logf(p);
  }
  if (cnt <= 1) return 0.f;
  ent = ent / (__logf((float)cnt) + EPSF);
  return fminf(fmaxf(ent, 0.f), 1.f);
}

__device__ inline float entropy_col(const float* col, size_t stride) {
  float m = NEG_INF; int cnt = 0;
  for (int d = 0; d < DP1; ++d) {
    float v = col[d * stride];
    if (v != NEG_INF) { cnt++; m = fmaxf(m, v); }
  }
  if (cnt == 0) return 0.f;
  float Z = 0.f;
  for (int d = 0; d < DP1; ++d)
    Z += __expf((col[d * stride] - m) * (1.0f / ENT_Tc));
  return entropy_finish(m, cnt, Z, col, stride);
}

__global__ void entropy_kernel(const float* __restrict__ E0, float* __restrict__ ent0) {
  const int idx = blockIdx.x * blockDim.x + threadIdx.x;
  if (idx >= Bn * H4c * W4c) return;
  const int b = idx / (H4c * W4c);
  const int rem = idx % (H4c * W4c);
  const int h = rem / W4c, x = rem % W4c;
  const size_t ds = (size_t)H4c * W4c;
  const float* col = E0 + ((size_t)b * DP1 * H4c + h) * W4c + x;
  ent0[idx] = entropy_col(col, ds);
}

// ============================================================================
// Refinement helpers: run extents + windowed mean over finite entries
// ============================================================================
__device__ inline void run_extents(const float* __restrict__ entrow, int x,
                                   int& a, int& bw) {
  int prev = -1;
  for (int j = x - 1; j >= 0; --j) if (entrow[j] <= ENT_THRc) { prev = j; break; }
  int next = W4c;
  for (int j = x + 1; j < W4c; ++j) if (entrow[j] <= ENT_THRc) { next = j; break; }
  a = prev + 1;               // == max(x-L,0) of reference
  int lo = x - WINH; if (lo > a) a = lo;
  bw = next - 1;              // == min(x+R, W4-1)
  int hiB = x + WINH; if (hiB < bw) bw = hiB;
}

__device__ inline float window_agg(const float* __restrict__ rowd, int a, int bw) {
  float num = 0.f, den = 0.f;
  for (int j = a; j <= bw; ++j) {
    float v = rowd[j];
    if (v != NEG_INF) { num += v; den += 1.f; }
  }
  return (den > 0.f) ? (num / den) : NEG_INF;
}

// ============================================================================
// Stage 1: E1 = refine(E0, ent0) at roi&invalid pixels; ent1 everywhere
// ============================================================================
__global__ void stage1_kernel(const float* __restrict__ E0,
                              const float* __restrict__ ent0,
                              float* __restrict__ E1,
                              float* __restrict__ ent1) {
  const int idx = blockIdx.x * blockDim.x + threadIdx.x;
  if (idx >= Bn * H4c * W4c) return;
  const int b = idx / (H4c * W4c);
  const int rem = idx % (H4c * W4c);
  const int h = rem / W4c, x = rem % W4c;
  const size_t ds = (size_t)H4c * W4c;
  const size_t colbase = ((size_t)b * DP1 * H4c + h) * W4c + x;

  const float* entrow = ent0 + (size_t)(b * H4c + h) * W4c;
  const float e = entrow[x];
  const bool invalid = (e > ENT_THRc);
  const bool roi = (x >= ROI_X0 && x <= ROI_X1 && h >= ROI_Y0 && h <= ROI_Y1);

  if (!(invalid && roi)) {                 // untouched column: copy through
    for (int d = 0; d < DP1; ++d)
      E1[colbase + (size_t)d * ds] = E0[colbase + (size_t)d * ds];
    ent1[idx] = e;
    return;
  }

  int a, bw;
  run_extents(entrow, x, a, bw);
  float col[DP1];
  for (int d = 0; d < DP1; ++d) {
    const float* rowd = E0 + ((size_t)(b * DP1 + d) * H4c + h) * W4c;
    col[d] = window_agg(rowd, a, bw);
    E1[colbase + (size_t)d * ds] = col[d];
  }
  ent1[idx] = entropy_col(col, 1);
}

// ============================================================================
// Stage 2 (fused): rm2 mask -> refine-on-the-fly argmax (=teacher) ->
// smooth-L1 vs student/4 -> per-sample atomic sums.  E2 never materialized:
// the loss only depends on teacher_cell at rm2 pixels.
// ============================================================================
__global__ void stage2_kernel(const float* __restrict__ E1,
                              const float* __restrict__ ent1,
                              const float* __restrict__ student,
                              float* __restrict__ sums,
                              float* __restrict__ cnts) {
  const int idx = blockIdx.x * blockDim.x + threadIdx.x;
  if (idx >= Bn * H4c * W4c) return;
  const int b = idx / (H4c * W4c);
  const int rem = idx % (H4c * W4c);
  const int h = rem / W4c, x = rem % W4c;

  const float* entrow = ent1 + (size_t)(b * H4c + h) * W4c;
  const bool invalid = (entrow[x] > ENT_THRc);
  const bool roi = (x >= ROI_X0 && x <= ROI_X1 && h >= ROI_Y0 && h <= ROI_Y1);
  if (!(invalid && roi)) return;           // mask==0 -> contributes nothing

  int a, bw;
  run_extents(entrow, x, a, bw);

  float best = NEG_INF; int bestd = 0;     // first-max == jnp.argmax
  for (int d = 0; d < DP1; ++d) {
    const float* rowd = E1 + ((size_t)(b * DP1 + d) * H4c + h) * W4c;
    float v = window_agg(rowd, a, bw);
    if (v > best) { best = v; bestd = d; }
  }

  const float teacher = (float)bestd;
  const float st = student[idx] * 0.25f;   // px -> cell units
  const float ad = fabsf(st - teacher);
  const float sl1 = (ad < 1.0f) ? (0.5f * ad * ad) : (ad - 0.5f);
  atomicAdd(&sums[b], sl1);
  atomicAdd(&cnts[b], 1.0f);
}

__global__ void init_kernel(float* __restrict__ sums, float* __restrict__ cnts) {
  if (threadIdx.x < Bn) { sums[threadIdx.x] = 0.f; cnts[threadIdx.x] = 0.f; }
}

__global__ void finalize_kernel(const float* __restrict__ sums,
                                const float* __restrict__ cnts,
                                float* __restrict__ out) {
  if (threadIdx.x == 0) {
    float acc = 0.f;
    for (int b = 0; b < Bn; ++b) acc += sums[b] / fmaxf(cnts[b], 1.0f);
    out[0] = acc * (1.0f / (float)Bn);
  }
}

// ============================================================================
extern "C" void kernel_launch(void* const* d_in, const int* in_sizes, int n_in,
                              void* d_out, int out_size, void* d_ws, size_t ws_size,
                              hipStream_t stream) {
  (void)in_sizes; (void)n_in; (void)out_size; (void)ws_size;
  const float* fL      = (const float*)d_in[0];
  const float* fR      = (const float*)d_in[1];
  const float* student = (const float*)d_in[2];
  float* out = (float*)d_out;

  float* ws = (float*)d_ws;
  const size_t N0 = (size_t)Bn * DP1 * H4c * W4c;  // cost volume elements
  const size_t NP = (size_t)Bn * H4c * W4c;        // pixels
  float* E0   = ws;
  float* E1   = E0 + N0;
  float* ent0 = E1 + N0;
  float* ent1 = ent0 + NP;
  float* sums = ent1 + NP;
  float* cnts = sums + Bn;

  // Allow the 249.6 KB dynamic-LDS request (host-side attribute set; not a
  // stream op, safe under graph capture; deterministic every call).
  (void)hipFuncSetAttribute((const void*)cv_wmma_row_kernel,
                            hipFuncAttributeMaxDynamicSharedMemorySize,
                            LDS_BYTES);

  init_kernel<<<1, 32, 0, stream>>>(sums, cnts);

  dim3 g1(H4c, Bn);                                // one row per workgroup
  cv_wmma_row_kernel<<<g1, 256, LDS_BYTES, stream>>>(fL, fR, E0);

  const int tb = 256;
  const int nb = (int)((NP + tb - 1) / tb);
  entropy_kernel<<<nb, tb, 0, stream>>>(E0, ent0);
  stage1_kernel<<<nb, tb, 0, stream>>>(E0, ent0, E1, ent1);
  stage2_kernel<<<nb, tb, 0, stream>>>(E1, ent1, student, sums, cnts);
  finalize_kernel<<<1, 32, 0, stream>>>(sums, cnts, out);
}
